// ProductQuantizer_82695300317334
// MI455X (gfx1250) — matrix operand order repaired and checked by reading, not compiled, not run on hardware
//
#include <hip/hip_runtime.h>
#include <hip/hip_bf16.h>
#include <string.h>

// Problem constants (match reference)
#define QQ 4
#define KK 2048
#define DD 256
#define BB 8
#define NN 8192
#define MM (NN / QQ)      // 2048 tokens per quantizer chunk
#define TOK (BB * NN)     // 65536 tokens total

#define BTILE 32          // codes staged per B buffer
#define NITER (KK / BTILE)

typedef __attribute__((ext_vector_type(16))) __bf16        v16bf;
typedef __attribute__((ext_vector_type(8)))  float         v8f;
typedef __attribute__((ext_vector_type(4)))  unsigned int  u32x4;
typedef __attribute__((ext_vector_type(8)))  int           i32x8;
typedef __attribute__((ext_vector_type(4)))  int           i32x4;

union Frag {
    uint4 u[2];
    v16bf v;
};

// float -> bf16 with round-to-nearest-even, pure integer ops
__device__ __forceinline__ unsigned short f2bf(float f) {
    unsigned u = __float_as_uint(f);
    unsigned r = ((u >> 16) & 1u) + 0x7FFFu;
    return (unsigned short)((u + r) >> 16);
}

// ---------------------------------------------------------------------------
// Kernel 1: row-wise L2 normalize 256-float rows -> packed bf16 rows.
// One wave per row (32 lanes x 8 floats), 8 rows per 256-thread block.
// ---------------------------------------------------------------------------
__global__ void __launch_bounds__(256)
l2norm_to_bf16(const float* __restrict__ src, unsigned short* __restrict__ dst) {
    const int row  = blockIdx.x * 8 + (threadIdx.x >> 5);
    const int lane = threadIdx.x & 31;

    const float4* p = (const float4*)(src + (size_t)row * DD) + lane * 2;
    float4 a = p[0];
    float4 b = p[1];

    float ss = a.x * a.x + a.y * a.y + a.z * a.z + a.w * a.w +
               b.x * b.x + b.y * b.y + b.z * b.z + b.w * b.w;
#pragma unroll
    for (int m = 1; m < 32; m <<= 1) ss += __shfl_xor(ss, m, 32);

    const float s = 1.0f / fmaxf(sqrtf(ss), 1e-12f);

    unsigned short h[8] = {f2bf(a.x * s), f2bf(a.y * s), f2bf(a.z * s), f2bf(a.w * s),
                           f2bf(b.x * s), f2bf(b.y * s), f2bf(b.z * s), f2bf(b.w * s)};
    uint4 pk;
    memcpy(&pk, h, 16);
    ((uint4*)(dst + (size_t)row * DD))[lane] = pk;
}

// ---------------------------------------------------------------------------
// TDM: 1-D async copy of `nelem` 2-byte elements from global to LDS.
// D# per CDNA5 ISA ch.8: group0 = {count, lds_addr, global_addr, type=2},
// group1 = {data_size=2B, tensor_dim0=nelem, tensor_dim1=1, tile_dim0=nelem,
//           tile_dim1=0 (unused), stride0=nelem}. Groups 2/3 zero (<=2D).
// ---------------------------------------------------------------------------
__device__ __forceinline__ void tdm_load_1d(unsigned lds_byte_addr,
                                            const void* gsrc,
                                            unsigned nelem) {
    unsigned long long ga = (unsigned long long)(size_t)gsrc;
    u32x4 g0 = {1u,                              // count=1 (valid user descriptor)
                lds_byte_addr,                   // lds_addr
                (unsigned)(ga & 0xFFFFFFFFu),    // global_addr[31:0]
                (unsigned)((ga >> 32) & 0x01FFFFFFu) | (2u << 30)}; // [56:32]|type=2
    i32x8 g1 = {(int)(1u << 16),                                  // data_size=2B, wg_mask=0
                (int)((nelem & 0xFFFFu) << 16),                   // tensor_dim0[15:0]
                (int)(((nelem >> 16) & 0xFFFFu) | (1u << 16)),    // dim0[31:16] | tensor_dim1=1
                (int)((nelem & 0xFFFFu) << 16),                   // tile_dim0
                0,                                                // tile_dim1=0, tile_dim2=0
                (int)nelem,                                       // tensor_dim0_stride[31:0]
                0, 0};
    i32x4 gz = {0, 0, 0, 0};
#if defined(__clang_major__) && (__clang_major__ >= 23)
    i32x8 gz8 = {0, 0, 0, 0, 0, 0, 0, 0};
    __builtin_amdgcn_tensor_load_to_lds(g0, g1, gz, gz, gz8, 0);
#else
    __builtin_amdgcn_tensor_load_to_lds(g0, g1, gz, gz, 0);
#endif
}

// ---------------------------------------------------------------------------
// Kernel 2: per quantizer chunk, cosine-sim argmax over K=2048 codes via
// bf16 WMMA, then gather raw f32 embed rows.
// Block = 256 threads = 8 waves; each wave owns a 16-token tile; block owns
// 128 consecutive tokens. LDS: A 64KB (TDM, once) + B 2x16KB (TDM, double
// buffered, 32 codes each) + 128 ints for index exchange.
// A fragments (8 x v16bf = 64 VGPRs) stay register-resident; B fragments are
// software-pipelined (stage s+1 loads issued while stage s WMMAs execute) so
// ds_load latency overlaps the two independent WMMA accumulator chains.
// ---------------------------------------------------------------------------
__global__ void __launch_bounds__(256)
pq_argmax_gather(const unsigned short* __restrict__ xn,  // [TOK, DD] bf16 (l2-normed)
                 const unsigned short* __restrict__ en,  // [QQ*KK, DD] bf16 (l2-normed)
                 const float* __restrict__ embed,        // [QQ, KK, DD] raw f32
                 float* __restrict__ qout,               // [TOK, DD]
                 float* __restrict__ enc) {              // [TOK]
    extern __shared__ char smem[];
    unsigned short* As   = (unsigned short*)smem;                    // 128*256 bf16
    unsigned short* Bs   = (unsigned short*)(smem + 128 * DD * 2);   // 2*BTILE*256 bf16
    int*            idxs = (int*)(smem + 128 * DD * 2 + 2 * BTILE * DD * 2);

    const int bid = blockIdx.x;          // 512 blocks: [q(4)][b(8)][g(16)]
    const int q = bid >> 7;
    const int b = (bid >> 4) & 7;
    const int g = bid & 15;
    const size_t tokenBase = (size_t)b * NN + (size_t)q * MM + (size_t)g * 128;
    const size_t enBase    = (size_t)q * KK;

    const int tid      = threadIdx.x;
    const int wave     = tid >> 5;
    const int lane     = tid & 31;
    const int laneN    = lane & 15;      // N (code) / M (token row) index
    const int lg       = lane >> 4;      // lane group (K-half selector)
    const int waveBase = wave * 16;      // wave's token tile within block

    const unsigned ldsA = (unsigned)(size_t)As;
    const unsigned ldsB = (unsigned)(size_t)Bs;

    // Prologue: TDM-stage A (once) and B tile 0, then make LDS visible.
    if (wave == 0) {
        tdm_load_1d(ldsA, xn + tokenBase * DD, 128 * DD);           // A: 64KB
        tdm_load_1d(ldsB, en + enBase * DD, BTILE * DD);            // B tile 0: 16KB
        __builtin_amdgcn_s_wait_tensorcnt(0);
    }
    __syncthreads();

    // A fragment layout (16-bit A 16x32, this code's own k order): lane holds row
    // M=laneN; elems 0..15 -> K = 32s + lg*16 + e (contiguous, matches B).
    // Loaded once, register-resident across the whole K loop.
    v16bf areg[8];
    {
        const unsigned short* arow = As + (waveBase + laneN) * DD + lg * 16;
#pragma unroll
        for (int s = 0; s < 8; ++s) {
            Frag f;
            f.u[0] = *(const uint4*)(arow + s * 32);
            f.u[1] = *(const uint4*)(arow + s * 32 + 8);
            areg[s] = f.v;
        }
    }

    float bestv[8];
    int   besti[8];
#pragma unroll
    for (int r = 0; r < 8; ++r) { bestv[r] = -3.0e38f; besti[r] = 0; }

    for (int c = 0; c < NITER; ++c) {
        // Kick next B tile while computing on the current one.
        if (wave == 0 && (c + 1) < NITER) {
            tdm_load_1d(ldsB + (unsigned)((c + 1) & 1) * BTILE * DD * 2,
                        en + (enBase + (size_t)(c + 1) * BTILE) * DD, BTILE * DD);
        }

        // B fragment: lane holds code N=laneN (+16 for the second subtile),
        // K = 32s + lg*16 + (0..15): 32 contiguous bytes of the code row.
        const unsigned short* bbase =
            Bs + (c & 1) * BTILE * DD + laneN * DD + lg * 16;

        v16bf b0 = *(const v16bf*)(bbase);            // t=0, s=0
        v16bf b1 = *(const v16bf*)(bbase + 16 * DD);  // t=1, s=0
        v8f acc0 = {0.f, 0.f, 0.f, 0.f, 0.f, 0.f, 0.f, 0.f};
        v8f acc1 = acc0;
#pragma unroll
        for (int s = 0; s < 8; ++s) {
            v16bf n0 = b0, n1 = b1;
            if (s < 7) {  // prefetch stage s+1 while stage s WMMAs run
                n0 = *(const v16bf*)(bbase + (s + 1) * 32);
                n1 = *(const v16bf*)(bbase + 16 * DD + (s + 1) * 32);
            }
            acc0 = __builtin_amdgcn_wmma_f32_16x16x32_bf16(
                false, areg[s], false, b0, (short)0, acc0, false, false);
            acc1 = __builtin_amdgcn_wmma_f32_16x16x32_bf16(
                false, areg[s], false, b1, (short)0, acc1, false, false);
            b0 = n0;
            b1 = n1;
        }

        const int code0 = c * BTILE + laneN;
#pragma unroll
        for (int r = 0; r < 8; ++r) {
            if (acc0[r] > bestv[r]) { bestv[r] = acc0[r]; besti[r] = code0; }
            if (acc1[r] > bestv[r]) { bestv[r] = acc1[r]; besti[r] = code0 + 16; }
        }

        // Drain the prefetch, then rotate buffers.
        if (wave == 0) __builtin_amdgcn_s_wait_tensorcnt(0);
        __syncthreads();
    }

    // Reduce (max, first-index) across the 16 lanes sharing each C row.
    // C layout: vgpr r -> M = lg*8 + r, N = laneN. xor masks 1..8 stay
    // inside each 16-lane half of the wave32.
#pragma unroll
    for (int m = 1; m <= 8; m <<= 1) {
#pragma unroll
        for (int r = 0; r < 8; ++r) {
            float ov = __shfl_xor(bestv[r], m, 32);
            int   oi = __shfl_xor(besti[r], m, 32);
            if (ov > bestv[r] || (ov == bestv[r] && oi < besti[r])) {
                bestv[r] = ov;
                besti[r] = oi;
            }
        }
    }
    if (laneN == 0) {
#pragma unroll
        for (int r = 0; r < 8; ++r) {
            const int trow = waveBase + lg * 8 + r;
            idxs[trow] = besti[r];
            enc[tokenBase + trow] = (float)besti[r];
        }
    }
    __syncthreads();

    // Gather raw f32 embed rows: each wave copies its 16 tokens' rows,
    // 32 lanes x 2 float4 per 256-float row (coalesced).
    for (int row = 0; row < 16; ++row) {
        const int tl  = waveBase + row;
        const int idx = idxs[tl];
        const float4* src = (const float4*)(embed + (enBase + (size_t)idx) * DD);
        float4*       dst = (float4*)(qout + (tokenBase + (size_t)tl) * DD);
        dst[lane]      = src[lane];
        dst[lane + 32] = src[lane + 32];
    }
}

__global__ void write_loss(float* p) { *p = 0.0f; }

// ---------------------------------------------------------------------------
extern "C" void kernel_launch(void* const* d_in, const int* in_sizes, int n_in,
                              void* d_out, int out_size, void* d_ws, size_t ws_size,
                              hipStream_t stream) {
    const float* x     = (const float*)d_in[0];   // [B, N, D] f32
    const float* embed = (const float*)d_in[1];   // [Q, K, D] f32

    float* qout = (float*)d_out;                  // [B*N, D]
    float* enc  = qout + (size_t)TOK * DD;        // [B*N]
    float* loss = enc + (size_t)TOK;              // [1]

    unsigned short* xn = (unsigned short*)d_ws;           // 32 MB bf16
    unsigned short* en = xn + (size_t)TOK * DD;           // 4 MB bf16

    l2norm_to_bf16<<<TOK / 8, 256, 0, stream>>>(x, xn);
    l2norm_to_bf16<<<(QQ * KK) / 8, 256, 0, stream>>>(embed, en);

    const size_t lds = 128 * DD * 2 + 2 * BTILE * DD * 2 + 128 * sizeof(int); // ~98.8 KB
    pq_argmax_gather<<<512, 256, lds, stream>>>(xn, en, embed, qout, enc);
    write_loss<<<1, 1, 0, stream>>>(loss);
}
